// Qwen3MoeOptSparseMoeBlock_76287209111680
// MI455X (gfx1250) — matrix operand order repaired and checked by reading, not compiled, not run on hardware
//
#include <hip/hip_runtime.h>
#include <hip/hip_bf16.h>

// Problem sizes (fixed by the reference)
#define TT 2048   // tokens (B*S)
#define HH 1024   // hidden dim
#define EE 16     // experts
#define II 512    // intermediate
#define KK 4      // top-k

#define MT 32     // token tile per block (2 x 16-row WMMA subtiles)
#define XSP 1032  // padded LDS stride for xs  (bank-conflict-free: +4 dwords)
#define ACTP 136  // padded LDS stride for act (bank-conflict-free: +4 dwords)

typedef __attribute__((ext_vector_type(16))) __bf16 v16bf;
typedef __attribute__((ext_vector_type(8)))  float  v8f;

static __device__ __forceinline__ __bf16 bfc(float x) { return (__bf16)x; }

// ---------------------------------------------------------------------------
// Kernel 1: router logits = x @ w_gate   ([T,H] x [H,E] -> [T,E])
// ---------------------------------------------------------------------------
__global__ void moe_router_logits(const float* __restrict__ x,
                                  const float* __restrict__ wg,
                                  float* __restrict__ logits) {
  int idx = blockIdx.x * blockDim.x + threadIdx.x;
  if (idx >= TT * EE) return;
  int t = idx >> 4, e = idx & 15;
  const float* xr = x + (size_t)t * HH;
  float s = 0.f;
  #pragma unroll 8
  for (int h = 0; h < HH; ++h) s = fmaf(xr[h], wg[h * EE + e], s);
  logits[t * EE + e] = s;
}

// ---------------------------------------------------------------------------
// Kernel 2: softmax + top-4 (ties -> lowest index, matching jax top_k) +
// renorm; scatter tokens into per-expert lists (integer atomics only).
// ---------------------------------------------------------------------------
__global__ void moe_route(const float* __restrict__ logits,
                          int* __restrict__ counts,
                          int* __restrict__ tok_idx,
                          int* __restrict__ tok_dest,
                          float* __restrict__ tok_w,
                          int mode) {
  int t = blockIdx.x * blockDim.x + threadIdx.x;
  if (t >= TT) return;
  float l[EE];
  float mx = -1e30f;
  #pragma unroll
  for (int e = 0; e < EE; ++e) { l[e] = logits[t * EE + e]; mx = fmaxf(mx, l[e]); }
  float sum = 0.f;
  #pragma unroll
  for (int e = 0; e < EE; ++e) { l[e] = __expf(l[e] - mx); sum += l[e]; }
  float inv = 1.f / sum;
  #pragma unroll
  for (int e = 0; e < EE; ++e) l[e] *= inv;

  int sel[KK]; float w[KK]; float wsum = 0.f;
  bool used[EE];
  #pragma unroll
  for (int e = 0; e < EE; ++e) used[e] = false;
  #pragma unroll
  for (int k = 0; k < KK; ++k) {
    float best = -1.f; int bi = 0;
    #pragma unroll
    for (int e = 0; e < EE; ++e)
      if (!used[e] && l[e] > best) { best = l[e]; bi = e; }
    used[bi] = true; sel[k] = bi; w[k] = best; wsum += best;
  }
  float winv = 1.f / wsum;   // NORM_TOPK_PROB
  for (int k = 0; k < KK; ++k) {
    int e = sel[k];
    int slot = atomicAdd(&counts[e], 1);
    tok_idx [e * TT + slot] = t;
    tok_dest[e * TT + slot] = mode ? t : (t * KK + k);
    tok_w   [e * TT + slot] = w[k] * winv;
  }
}

// ---------------------------------------------------------------------------
// WMMA operand loaders (CDNA5 16-bit layouts, wave32).
// A (16xK bf16): lane l -> row row0+(l&15); vector elems 0..7 = K kb..kb+7,
//                elems 8..15 = K kb+16..kb+23, kb = k0 + (l>=16 ? 8 : 0).
//                Two aligned 16B LDS reads (ds_read_b128).
// B (Kx16 bf16): lane l -> column col0+(l&15); elems 0..7 = K kb..kb+7,
//                elems 8..15 = K kb+16..kb+23, kb = k0 + (l>=16 ? 8 : 0).
// ---------------------------------------------------------------------------
union V16 { v16bf v; unsigned int u[8]; };

static __device__ __forceinline__ v16bf load_a16(const __bf16* base,
                                                 int rowStride, int lane,
                                                 int row0, int k0) {
  int row = row0 + (lane & 15);
  int kb  = k0 + ((lane & 16) ? 8 : 0);
  const uint4* p = (const uint4*)(base + row * rowStride + kb);
  uint4 lo = p[0];   // K = kb .. kb+7
  uint4 hi = p[2];   // K = kb+16 .. kb+23
  V16 r;
  r.u[0] = lo.x; r.u[1] = lo.y; r.u[2] = lo.z; r.u[3] = lo.w;
  r.u[4] = hi.x; r.u[5] = hi.y; r.u[6] = hi.z; r.u[7] = hi.w;
  return r.v;
}

static __device__ __forceinline__ v16bf load_b16(const float* __restrict__ base,
                                                 int rowStride, int lane,
                                                 int k0, int col0) {
  const int kb = k0 + ((lane & 16) ? 8 : 0);
  const float* p = base + (size_t)kb * rowStride + col0 + (lane & 15);
  V16 r;
  #pragma unroll
  for (int j = 0; j < 8; ++j) {
    r.v[j]     = bfc(p[(size_t)j * rowStride]);
    r.v[8 + j] = bfc(p[(size_t)(16 + j) * rowStride]);
  }
  return r.v;
}

// ---------------------------------------------------------------------------
// Kernel 3: per-expert fused MLP.
// Grid: (T/32 token tiles, E experts). Block: 256 threads = 8 waves.
// M=32 token tile (2 WMMA subtiles) -> each weight B-tile feeds 2 WMMAs and
// per-block weight traffic halves vs M=16. Each wave owns a 128-wide H slice.
// ---------------------------------------------------------------------------
__global__ __launch_bounds__(256, 1) void moe_expert_gemm(
    const float* __restrict__ x,
    const float* __restrict__ w_gate_proj,   // [E][H][I]
    const float* __restrict__ w_up_proj,     // [E][H][I]
    const float* __restrict__ w_down_proj,   // [E][I][H]
    const int*   __restrict__ counts,
    const int*   __restrict__ tok_idx,
    const int*   __restrict__ tok_dest,
    const float* __restrict__ tok_w,
    float* __restrict__ dstbuf,              // mode0: contrib[T*4][H]; mode1: out[T][H]
    int mode) {
  __shared__ __bf16 xs[MT][XSP];     // ~66 KB token activations (bf16, padded)
  __shared__ __bf16 act[MT][ACTP];   // ~8.7 KB intermediate tile (padded)
  __shared__ int    dstM[MT];
  __shared__ float  wM[MT];

  const int e   = blockIdx.y;
  const int n_e = counts[e];
  const int m0  = blockIdx.x * MT;
  if (m0 >= n_e) return;             // uniform per block

  const int tid  = threadIdx.x;
  const int lane = tid & 31;
  const int wave = tid >> 5;

  // Stage token rows into LDS (fp32 -> bf16), zero-fill ragged tail.
  for (int r = 0; r < MT; ++r) {
    int li = m0 + r;
    if (li < n_e) {
      const float* src = x + (size_t)tok_idx[e * TT + li] * HH;
      for (int c = tid; c < HH; c += 256) xs[r][c] = bfc(src[c]);
      if (tid == 0) { dstM[r] = tok_dest[e * TT + li]; wM[r] = tok_w[e * TT + li]; }
    } else {
      for (int c = tid; c < HH; c += 256) xs[r][c] = (__bf16)0.f;
      if (tid == 0) { dstM[r] = -1; wM[r] = 0.f; }
    }
  }
  __syncthreads();

  const float* wg_e = w_gate_proj + (size_t)e * HH * II;
  const float* wu_e = w_up_proj   + (size_t)e * HH * II;
  const float* wd_e = w_down_proj + (size_t)e * II * HH;
  const int hbase = wave * 128;      // this wave's H-slice

  v8f accO[2][8];
  #pragma unroll
  for (int ms = 0; ms < 2; ++ms)
    #pragma unroll
    for (int i = 0; i < 8; ++i)
      #pragma unroll
      for (int j = 0; j < 8; ++j) accO[ms][i][j] = 0.f;

  const int nloc = lane & 15;
  const int mhi  = (lane & 16) ? 8 : 0;

  for (int i0 = 0; i0 < II; i0 += 128) {
    // ---- gate/up: this wave's 16 intermediate cols x 32 tokens, K = 1024
    const int col0 = i0 + wave * 16;
    v8f accG[2], accU[2];
    #pragma unroll
    for (int ms = 0; ms < 2; ++ms)
      #pragma unroll
      for (int j = 0; j < 8; ++j) { accG[ms][j] = 0.f; accU[ms][j] = 0.f; }

    for (int k0 = 0; k0 < HH; k0 += 32) {
      v16bf bg = load_b16(wg_e, II, lane, k0, col0);
      v16bf bu = load_b16(wu_e, II, lane, k0, col0);
      if (k0 + 32 < HH) {  // prefetch next K-chunk rows (global_prefetch_b8)
        __builtin_prefetch((const void*)(wg_e + (size_t)(k0 + 32 + lane) * II + col0), 0, 0);
        __builtin_prefetch((const void*)(wu_e + (size_t)(k0 + 32 + lane) * II + col0), 0, 0);
      }
      #pragma unroll
      for (int ms = 0; ms < 2; ++ms) {
        v16bf a = load_a16(&xs[0][0], XSP, lane, ms * 16, k0);
        accG[ms] = __builtin_amdgcn_wmma_f32_16x16x32_bf16(false, a, false, bg,
                                                           (short)0, accG[ms], false, false);
        accU[ms] = __builtin_amdgcn_wmma_f32_16x16x32_bf16(false, a, false, bu,
                                                           (short)0, accU[ms], false, false);
      }
    }
    __syncthreads();   // previous chunk's act fully consumed by all waves
    #pragma unroll
    for (int ms = 0; ms < 2; ++ms)
      #pragma unroll
      for (int r = 0; r < 8; ++r) {
        float g = accG[ms][r];
        float s = g * __builtin_amdgcn_rcpf(1.f + __expf(-g));   // fast silu
        act[ms * 16 + r + mhi][wave * 16 + nloc] = bfc(s * accU[ms][r]);
      }
    __syncthreads();

    // ---- down projection: accO += act(32x128) @ Wd[i0:i0+128, hbase:+128]
    for (int kc = 0; kc < 128; kc += 32) {
      v16bf a0 = load_a16(&act[0][0], ACTP, lane, 0,  kc);
      v16bf a1 = load_a16(&act[0][0], ACTP, lane, 16, kc);
      if (kc + 32 < 128)
        __builtin_prefetch((const void*)(wd_e + (size_t)(i0 + kc + 32 + lane) * HH + hbase), 0, 0);
      #pragma unroll
      for (int nt = 0; nt < 8; ++nt) {
        v16bf bd = load_b16(wd_e, HH, lane, i0 + kc, hbase + nt * 16);
        accO[0][nt] = __builtin_amdgcn_wmma_f32_16x16x32_bf16(false, a0, false, bd,
                                                              (short)0, accO[0][nt], false, false);
        accO[1][nt] = __builtin_amdgcn_wmma_f32_16x16x32_bf16(false, a1, false, bd,
                                                              (short)0, accO[1][nt], false, false);
      }
    }
  }

  // ---- scale by routing weight, write out. D layout: VGPR r -> row r+mhi,
  // lane%16 -> column.
  #pragma unroll
  for (int ms = 0; ms < 2; ++ms)
    #pragma unroll
    for (int r = 0; r < 8; ++r) {
      int m = ms * 16 + r + mhi;
      int d = dstM[m];
      if (d < 0) continue;
      float w = wM[m];
      float* orow = dstbuf + (size_t)d * HH + hbase;
      #pragma unroll
      for (int nt = 0; nt < 8; ++nt) {
        float val = accO[ms][nt][r] * w;
        if (mode) atomicAdd(&orow[nt * 16 + nloc], val);
        else      orow[nt * 16 + nloc] = val;
      }
    }
}

// ---------------------------------------------------------------------------
// Kernel 4 (mode 0): deterministic fixed-order combine of the 4 slots.
// ---------------------------------------------------------------------------
__global__ void moe_combine(const float* __restrict__ contrib,
                            float* __restrict__ out) {
  int idx = blockIdx.x * blockDim.x + threadIdx.x;
  if (idx >= TT * HH) return;
  int t = idx / HH, h = idx - t * HH;
  const float* c = contrib + ((size_t)t * KK) * HH + h;
  float s = c[0];
  s += c[(size_t)1 * HH];
  s += c[(size_t)2 * HH];
  s += c[(size_t)3 * HH];
  out[idx] = s;
}

// ---------------------------------------------------------------------------
extern "C" void kernel_launch(void* const* d_in, const int* in_sizes, int n_in,
                              void* d_out, int out_size, void* d_ws, size_t ws_size,
                              hipStream_t stream) {
  (void)in_sizes; (void)n_in; (void)out_size;
  const float* x   = (const float*)d_in[0];
  const float* wg  = (const float*)d_in[1];
  const float* wgp = (const float*)d_in[2];
  const float* wup = (const float*)d_in[3];
  const float* wdp = (const float*)d_in[4];

  float* out    = (float*)d_out;                    // [T*H] then [T*E]
  float* logits = out + (size_t)TT * HH;

  char*  ws  = (char*)d_ws;
  int*   counts   = (int*)ws;
  size_t off      = 256;
  int*   tok_idx  = (int*)(ws + off); off += (size_t)EE * TT * sizeof(int);
  int*   tok_dest = (int*)(ws + off); off += (size_t)EE * TT * sizeof(int);
  float* tok_w    = (float*)(ws + off); off += (size_t)EE * TT * sizeof(float);
  size_t contrib_need = (size_t)TT * KK * HH * sizeof(float);   // 32 MB
  int    mode = (ws_size >= off + contrib_need) ? 0 : 1;        // 1 = atomic fallback
  float* contrib = (mode == 0) ? (float*)(ws + off) : out;

  hipMemsetAsync(counts, 0, 64, stream);
  if (mode == 1)
    hipMemsetAsync(out, 0, (size_t)TT * HH * sizeof(float), stream);

  moe_router_logits<<<(TT * EE + 255) / 256, 256, 0, stream>>>(x, wg, logits);
  moe_route<<<(TT + 255) / 256, 256, 0, stream>>>(logits, counts, tok_idx,
                                                  tok_dest, tok_w, mode);
  dim3 grid(TT / MT, EE);
  moe_expert_gemm<<<grid, 256, 0, stream>>>(x, wgp, wup, wdp, counts, tok_idx,
                                            tok_dest, tok_w, contrib, mode);
  if (mode == 0)
    moe_combine<<<(TT * HH + 255) / 256, 256, 0, stream>>>(contrib, out);
}
